// VQEmbeddingEMA_Predictive_DDP_83124797047430
// MI455X (gfx1250) — matrix operand order repaired and checked by reading, not compiled, not run on hardware
//
#include <hip/hip_runtime.h>
#include <hip/hip_bf16.h>
#include <math.h>

// Problem constants
#define B_   16
#define T_   2048
#define D_   256
#define M_   1024
#define K_   7
#define TC_  2054   // T-1+K
#define TCP_ 2064   // padded row count for ctx_noisy (transposed layout [b][j][i])
#define NTOK (B_*T_)
#define NELEM (B_*T_*D_)
#define PITCH 40    // LDS tile pitch in ushorts (80 bytes)

typedef __attribute__((ext_vector_type(16))) __bf16         v16bf;
typedef __attribute__((ext_vector_type(16))) unsigned short us16;
typedef __attribute__((ext_vector_type(8)))  unsigned short us8;
typedef __attribute__((ext_vector_type(8)))  float          v8f;
typedef __attribute__((ext_vector_type(4)))  unsigned int   u32x4;
typedef __attribute__((ext_vector_type(8)))  unsigned int   u32x8;

__device__ __forceinline__ unsigned short f2bf(float f) {
  unsigned int u = __builtin_bit_cast(unsigned int, f);
  unsigned int r = 0x7FFFu + ((u >> 16) & 1u);
  return (unsigned short)((u + r) >> 16);
}

// Two 16-byte LDS loads -> one 16-element bf16 fragment
__device__ __forceinline__ v16bf load_frag(const unsigned short* p0, const unsigned short* p1) {
  us8 lo = *(const us8*)p0;
  us8 hi = *(const us8*)p1;
  us16 v = __builtin_shufflevector(lo, hi, 0,1,2,3,4,5,6,7,8,9,10,11,12,13,14,15);
  return __builtin_bit_cast(v16bf, v);
}

// One K=32 step: wave computes 32(M) x 64(N) with 8 wmma (fragments preloaded).
// lds_a : [128][PITCH] (row = local M, col = local K)
// lds_bT: [ 64][PITCH] (row = local N, col = local K)  -- B stored N-major
__device__ __forceinline__ void wmma_tile2(const unsigned short* lds_a,
                                           const unsigned short* lds_bT,
                                           int aRow0, int lane, v8f acc[2][4]) {
  const int m     = lane & 15;
  const int kaddA = (lane < 16) ? 0 : 8;   // 16-bit A-matrix lane layout
  const int koff  = (lane < 16) ? 0 : 8;   // 16-bit B-matrix lane layout
  const int nn    = lane & 15;
  const unsigned short* pa0 = lds_a + (aRow0 + m) * PITCH + kaddA;
  const unsigned short* pa1 = pa0 + 16 * PITCH;
  v16bf a0 = load_frag(pa0, pa0 + 16);
  v16bf a1 = load_frag(pa1, pa1 + 16);
  v16bf bf[4];
#pragma unroll
  for (int nb = 0; nb < 4; ++nb) {
    const unsigned short* pb = lds_bT + (nb * 16 + nn) * PITCH + koff;
    bf[nb] = load_frag(pb, pb + 16);
  }
#pragma unroll
  for (int nb = 0; nb < 4; ++nb) {
    acc[0][nb] = __builtin_amdgcn_wmma_f32_16x16x32_bf16(
        false, a0, false, bf[nb], (short)0, acc[0][nb], false, false);
    acc[1][nb] = __builtin_amdgcn_wmma_f32_16x16x32_bf16(
        false, a1, false, bf[nb], (short)0, acc[1][nb], false, false);
  }
}

// ---------------- Tensor Data Mover helper ----------------
// 2-D tile load: `rows` rows x 32 bf16 (contiguous), row stride `stride` elements,
// LDS pitch padded to 80B (pad 16B after each 64B row).
__device__ __forceinline__ void tdm_load_tile_2d(unsigned lds_byte_off,
                                                 const unsigned short* gptr,
                                                 unsigned rows, unsigned stride) {
  unsigned long long ga = (unsigned long long)(const void*)gptr;
  u32x4 g0;
  g0[0] = 1u;                                   // count=1 (valid user descriptor)
  g0[1] = lds_byte_off;                         // LDS destination (bytes)
  g0[2] = (unsigned)ga;                         // global_addr lo
  g0[3] = (unsigned)(ga >> 32) | (2u << 30);    // global_addr hi | type=2 (image)
  u32x8 g1;
  g1[0] = (1u << 16)            // data_size = 2 bytes
        | (1u << 20)            // pad_enable
        | (3u << 22)            // pad_interval: 16 DWORDs (=64B, one tile row)
        | (3u << 25);           // pad_amount: 4 DWORDs (=16B) -> 80B LDS pitch
  g1[1] = 0u;                                   // barrier addr=0, tensor_dim0 lo16=0
  g1[2] = 0x10u;                                // tensor_dim0 = 1<<20 (no OOB clip)
  g1[3] = (32u << 16) | 0x10u;                  // tile_dim0=32 | tensor_dim1 = 1<<20
  g1[4] = rows;                                 // tile_dim1=rows, tile_dim2=0
  g1[5] = stride;                               // tensor_dim0_stride (elements)
  g1[6] = 0u;
  g1[7] = 0u;
  asm volatile("tensor_load_to_lds %0, %1" :: "s"(g0), "s"(g1) : "memory");
}

// ---------------- small prep kernels ----------------

__global__ void init_kernel(int* counts, float* sums) {
  int i = blockIdx.x * blockDim.x + threadIdx.x;
  if (i < M_) counts[i] = 0;
  if (i < 2)  sums[i] = 0.0f;
}

// generic f32 -> bf16 copy
__global__ void tobf_kernel(const float* __restrict__ src, unsigned short* __restrict__ dst, int n) {
  int i = blockIdx.x * blockDim.x + threadIdx.x;
  if (i < n) dst[i] = f2bf(src[i]);
}

// scale[b] = sqrt( sum(x[b,0:T-1,:]^2) / (D*TC) ) ; one block per batch
__global__ void scale_kernel(const float* __restrict__ x, float* __restrict__ scale) {
  __shared__ float red[256];
  int b = blockIdx.x, tid = threadIdx.x;
  const float* base = x + (size_t)b * T_ * D_;
  float s = 0.f;
  const int n = (T_ - 1) * D_;
  for (int i = tid; i < n; i += 256) { float v = base[i]; s += v * v; }
  red[tid] = s; __syncthreads();
  for (int o = 128; o > 0; o >>= 1) { if (tid < o) red[tid] += red[tid + o]; __syncthreads(); }
  if (tid == 0) scale[b] = sqrtf(red[0] / (float)(D_ * TC_));
}

// ctx_noisy TRANSPOSED (bf16) [B][TCP][D]: row j, feature i contiguous
__global__ void noisy_kernel(const float* __restrict__ x, const float* __restrict__ noise,
                             const float* __restrict__ scale, const int* __restrict__ epo,
                             unsigned short* __restrict__ ctxnT) {
  int i = blockIdx.x * blockDim.x + threadIdx.x;
  const int total = B_ * TCP_ * D_;
  if (i >= total) return;
  int dd = i & (D_ - 1);
  int rest = i >> 8;
  int j = rest % TCP_;
  int b = rest / TCP_;
  unsigned short out = 0;
  if (j < TC_) {
    float coef = 0.5f * exp2f(-(float)epo[0] * 0.1f);
    float base = (j >= K_) ? x[((size_t)b * T_ + (j - K_)) * D_ + dd] : 0.f;
    float nz   = noise[((size_t)b * D_ + dd) * TC_ + j];
    out = f2bf(base + coef * scale[b] * nz);
  }
  ctxnT[i] = out;
}

// repack w_ctx [O][I][7] f32 -> wctxr [7][O][I] bf16 (I contiguous)
__global__ void pack_wctx_kernel(const float* __restrict__ wctx, unsigned short* __restrict__ wctxr) {
  int i = blockIdx.x * blockDim.x + threadIdx.x;
  if (i >= K_ * D_ * D_) return;
  int kk = i / (D_ * D_);
  int rem = i % (D_ * D_);
  int o = rem / D_, ii = rem % D_;
  wctxr[i] = f2bf(wctx[((size_t)o * D_ + ii) * K_ + kk]);
}

__global__ void embsq_kernel(const float* __restrict__ emb, float* __restrict__ embsq) {
  int m = blockIdx.x * blockDim.x + threadIdx.x;
  if (m >= M_) return;
  float s = 0.f;
  for (int d = 0; d < D_; ++d) { float e = emb[(size_t)m * D_ + d]; s += e * e; }
  embsq[m] = s;
}

// ---------------- WMMA GEMM kernels (all TDM-fed, double-buffered) ----------------

// ctx_out[b,o,t] = sum_{i,kk} ctx_noisy[b,i,t+kk] * w_ctx[o,i,kk]
__global__ __launch_bounds__(128) void ctx_conv_kernel(const unsigned short* __restrict__ ctxnT,
                                                       const unsigned short* __restrict__ wctxr,
                                                       float* __restrict__ ctx_out,
                                                       unsigned short* __restrict__ ctxT) {
  __shared__ unsigned short lds_a[2][128 * PITCH];
  __shared__ unsigned short lds_b[2][64 * PITCH];
  const int b = blockIdx.x, o0 = blockIdx.y * 128, t0 = blockIdx.z * 64;
  const int tid = threadIdx.x, wave = tid >> 5, lane = tid & 31;
  v8f acc[2][4] = {};
  const int NSTEP = 56;   // 7 taps * 8 chunks of 32 over D

  auto issue = [&](int s, int buf) {
    int kk = s >> 3, i0 = (s & 7) << 5;
    tdm_load_tile_2d((unsigned)(size_t)&lds_a[buf][0],
                     wctxr + ((size_t)kk * D_ + o0) * D_ + i0, 128, D_);
    tdm_load_tile_2d((unsigned)(size_t)&lds_b[buf][0],
                     ctxnT + ((size_t)b * TCP_ + t0 + kk) * D_ + i0, 64, D_);
  };

  if (wave == 0) { issue(0, 0); __builtin_amdgcn_s_wait_tensorcnt(0); }
  __syncthreads();
  for (int s = 0; s < NSTEP; ++s) {
    const int buf = s & 1;
    if (wave == 0 && s + 1 < NSTEP) issue(s + 1, buf ^ 1);
    wmma_tile2(&lds_a[buf][0], &lds_b[buf][0], wave * 32, lane, acc);
    if (wave == 0 && s + 1 < NSTEP) __builtin_amdgcn_s_wait_tensorcnt(0);
    __syncthreads();
  }

  const int n = lane & 15, rowadd = (lane < 16) ? 0 : 8;
#pragma unroll
  for (int ma = 0; ma < 2; ++ma)
#pragma unroll
    for (int nb = 0; nb < 4; ++nb)
#pragma unroll
      for (int r = 0; r < 8; ++r) {
        int o = o0 + wave * 32 + ma * 16 + r + rowadd;
        int t = t0 + nb * 16 + n;
        float v = acc[ma][nb][r];
        ctx_out[((size_t)b * D_ + o) * T_ + t] = v;        // f32 [b][o][t] for losses
        ctxT[((size_t)b * T_ + t) * D_ + o]   = f2bf(v);   // bf16 token-major for fuse GEMMs
      }
}

// cur_x (bf16) [token][D] = prelu( w1 * [x^T ; ctx_out], a1 )
__global__ __launch_bounds__(128) void fuse1_kernel(const unsigned short* __restrict__ xbf,
                                                    const unsigned short* __restrict__ w1bf,
                                                    const float* __restrict__ a1p,
                                                    const unsigned short* __restrict__ ctxT,
                                                    unsigned short* __restrict__ curx) {
  __shared__ unsigned short lds_a[2][128 * PITCH];
  __shared__ unsigned short lds_b[2][64 * PITCH];
  const int b = blockIdx.x, o0 = blockIdx.y * 128, t0 = blockIdx.z * 64;
  const int tid = threadIdx.x, wave = tid >> 5, lane = tid & 31;
  v8f acc[2][4] = {};
  const int NSTEP = 16;   // K = 512 in chunks of 32

  auto issue = [&](int s, int buf) {
    int kb = s << 5;
    tdm_load_tile_2d((unsigned)(size_t)&lds_a[buf][0],
                     w1bf + (size_t)o0 * (2 * D_) + kb, 128, 2 * D_);
    const unsigned short* src = (kb < D_)
        ? (xbf  + ((size_t)b * T_ + t0) * D_ + kb)
        : (ctxT + ((size_t)b * T_ + t0) * D_ + (kb - D_));
    tdm_load_tile_2d((unsigned)(size_t)&lds_b[buf][0], src, 64, D_);
  };

  if (wave == 0) { issue(0, 0); __builtin_amdgcn_s_wait_tensorcnt(0); }
  __syncthreads();
  for (int s = 0; s < NSTEP; ++s) {
    const int buf = s & 1;
    if (wave == 0 && s + 1 < NSTEP) issue(s + 1, buf ^ 1);
    wmma_tile2(&lds_a[buf][0], &lds_b[buf][0], wave * 32, lane, acc);
    if (wave == 0 && s + 1 < NSTEP) __builtin_amdgcn_s_wait_tensorcnt(0);
    __syncthreads();
  }

  const float a1 = a1p[0];
  const int n = lane & 15, rowadd = (lane < 16) ? 0 : 8;
#pragma unroll
  for (int ma = 0; ma < 2; ++ma)
#pragma unroll
    for (int nb = 0; nb < 4; ++nb)
#pragma unroll
      for (int r = 0; r < 8; ++r) {
        float v = acc[ma][nb][r];
        v = (v >= 0.f) ? v : a1 * v;
        int tok = b * T_ + t0 + nb * 16 + n;
        int o = o0 + wave * 32 + ma * 16 + r + rowadd;
        curx[(size_t)tok * D_ + o] = f2bf(v);
      }
}

// argmin_m ( |e_m|^2 - 2 * cx . e_m ) via WMMA; writes indices + histogram
__global__ __launch_bounds__(128) void vq_kernel(const unsigned short* __restrict__ curx,
                                                 const unsigned short* __restrict__ embbf,
                                                 const float* __restrict__ embsq,
                                                 int* __restrict__ indices,
                                                 int* __restrict__ counts) {
  __shared__ unsigned short lds_a[2][128 * PITCH];
  __shared__ unsigned short lds_b[2][64 * PITCH];
  __shared__ float red_val[128 * 16];
  __shared__ int   red_idx[128 * 16];
  const int tok0 = blockIdx.x * 128;
  const int tid = threadIdx.x, wave = tid >> 5, lane = tid & 31;
  const int n = lane & 15;
  const int NSTEP = 128;  // 16 code blocks of 64 x 8 K-chunks of 32
  float bval[2][8]; int bidx[2][8];
#pragma unroll
  for (int ma = 0; ma < 2; ++ma)
#pragma unroll
    for (int r = 0; r < 8; ++r) { bval[ma][r] = 3.4e38f; bidx[ma][r] = 0; }

  auto issue = [&](int s, int buf) {
    int c0 = (s >> 3) << 6, kb = (s & 7) << 5;
    tdm_load_tile_2d((unsigned)(size_t)&lds_a[buf][0],
                     curx + (size_t)tok0 * D_ + kb, 128, D_);
    tdm_load_tile_2d((unsigned)(size_t)&lds_b[buf][0],
                     embbf + (size_t)c0 * D_ + kb, 64, D_);
  };

  v8f acc[2][4] = {};
  if (wave == 0) { issue(0, 0); __builtin_amdgcn_s_wait_tensorcnt(0); }
  __syncthreads();
  for (int s = 0; s < NSTEP; ++s) {
    const int buf = s & 1;
    if (wave == 0 && s + 1 < NSTEP) issue(s + 1, buf ^ 1);
    wmma_tile2(&lds_a[buf][0], &lds_b[buf][0], wave * 32, lane, acc);
    if (wave == 0 && s + 1 < NSTEP) __builtin_amdgcn_s_wait_tensorcnt(0);
    __syncthreads();
    if ((s & 7) == 7) {          // finished one 64-code block: fold into argmin
      int c0 = (s >> 3) << 6;
#pragma unroll
      for (int ma = 0; ma < 2; ++ma)
#pragma unroll
        for (int nb = 0; nb < 4; ++nb)
#pragma unroll
          for (int r = 0; r < 8; ++r) {
            int code = c0 + nb * 16 + n;
            float d = embsq[code] - 2.0f * acc[ma][nb][r];
            if (d < bval[ma][r]) { bval[ma][r] = d; bidx[ma][r] = code; }
            acc[ma][nb][r] = 0.f;
          }
    }
  }

  const int half = (lane < 16) ? 0 : 1;
#pragma unroll
  for (int ma = 0; ma < 2; ++ma)
#pragma unroll
    for (int r = 0; r < 8; ++r) {
      int tl = wave * 32 + ma * 16 + half * 8 + r;   // local token
      red_val[tl * 16 + n] = bval[ma][r];
      red_idx[tl * 16 + n] = bidx[ma][r];
    }
  __syncthreads();
  if (tid < 128) {
    float bv = red_val[tid * 16]; int bi = red_idx[tid * 16];
    for (int l = 1; l < 16; ++l) {
      float v = red_val[tid * 16 + l];
      if (v < bv) { bv = v; bi = red_idx[tid * 16 + l]; }
    }
    indices[tok0 + tid] = bi;
    atomicAdd(&counts[bi], 1);
  }
}

// q_output = prelu( w2 * [quantized^T ; ctx_out], a2 ), quantized gathered from embbf[indices]
__global__ __launch_bounds__(128) void fuse2_kernel(const unsigned short* __restrict__ embbf,
                                                    const unsigned short* __restrict__ w2bf,
                                                    const float* __restrict__ a2p,
                                                    const unsigned short* __restrict__ ctxT,
                                                    const int* __restrict__ indices,
                                                    float* __restrict__ out) {
  __shared__ unsigned short lds_a[2][128 * PITCH];
  __shared__ unsigned short lds_b[2][64 * PITCH];
  const int b = blockIdx.x, o0 = blockIdx.y * 128, t0 = blockIdx.z * 64;
  const int tid = threadIdx.x, wave = tid >> 5, lane = tid & 31;
  v8f acc[2][4] = {};
  const int NSTEP = 16;

  auto issueA = [&](int s, int buf) {
    int kb = s << 5;
    tdm_load_tile_2d((unsigned)(size_t)&lds_a[buf][0],
                     w2bf + (size_t)o0 * (2 * D_) + kb, 128, 2 * D_);
  };
  auto stageB = [&](int s, int buf) {
    int kb = s << 5;
    if (kb < D_) {       // data-dependent gather: manual stage from embbf
      for (int q = tid; q < 256; q += 128) {
        int n2 = q >> 2, k8 = (q & 3) * 8;
        int id = indices[b * T_ + t0 + n2];
        *(us8*)(&lds_b[buf][0] + n2 * PITCH + k8) =
            *(const us8*)(embbf + (size_t)id * D_ + kb + k8);
      }
    } else if (wave == 0) {
      tdm_load_tile_2d((unsigned)(size_t)&lds_b[buf][0],
                       ctxT + ((size_t)b * T_ + t0) * D_ + (kb - D_), 64, D_);
    }
  };

  if (wave == 0) issueA(0, 0);
  stageB(0, 0);
  if (wave == 0) __builtin_amdgcn_s_wait_tensorcnt(0);
  __syncthreads();
  for (int s = 0; s < NSTEP; ++s) {
    const int buf = s & 1;
    if (s + 1 < NSTEP) {
      if (wave == 0) issueA(s + 1, buf ^ 1);
      stageB(s + 1, buf ^ 1);
    }
    wmma_tile2(&lds_a[buf][0], &lds_b[buf][0], wave * 32, lane, acc);
    if (wave == 0 && s + 1 < NSTEP) __builtin_amdgcn_s_wait_tensorcnt(0);
    __syncthreads();
  }

  const float a2 = a2p[0];
  const int n = lane & 15, rowadd = (lane < 16) ? 0 : 8;
#pragma unroll
  for (int ma = 0; ma < 2; ++ma)
#pragma unroll
    for (int nb = 0; nb < 4; ++nb)
#pragma unroll
      for (int r = 0; r < 8; ++r) {
        float v = acc[ma][nb][r];
        v = (v >= 0.f) ? v : a2 * v;
        int t = t0 + nb * 16 + n;
        int o = o0 + wave * 32 + ma * 16 + r + rowadd;
        out[((size_t)b * T_ + t) * D_ + o] = v;
      }
}

// ---------------- epilogue kernels ----------------

__global__ void loss_kernel(const float* __restrict__ x, const float* __restrict__ ctx_out,
                            const float* __restrict__ emb, const int* __restrict__ indices,
                            float* __restrict__ sums) {
  __shared__ float sA[256], sB[256];
  int tid = threadIdx.x;
  size_t i = (size_t)blockIdx.x * blockDim.x + tid;
  size_t stride = (size_t)gridDim.x * blockDim.x;
  float cs = 0.f, ps = 0.f;
  for (; i < (size_t)NELEM; i += stride) {
    int dd = (int)(i & (D_ - 1));
    size_t bt = i >> 8;
    int t = (int)(bt & (T_ - 1));
    int b = (int)(bt >> 11);
    float xv = x[i];
    float cv = ctx_out[((size_t)b * D_ + dd) * T_ + t];
    float qv = emb[(size_t)indices[bt] * D_ + dd];
    float e = xv - cv - qv; cs += e * e;
    float p = cv - xv;      ps += p * p;
  }
  sA[tid] = cs; sB[tid] = ps; __syncthreads();
  for (int o = 128; o > 0; o >>= 1) {
    if (tid < o) { sA[tid] += sA[tid + o]; sB[tid] += sB[tid + o]; }
    __syncthreads();
  }
  if (tid == 0) { atomicAdd(&sums[0], sA[0]); atomicAdd(&sums[1], sB[0]); }
}

__global__ void idx2f_kernel(const int* __restrict__ indices, float* __restrict__ out) {
  int i = blockIdx.x * blockDim.x + threadIdx.x;
  if (i < NTOK) out[i] = (float)indices[i];
}

__global__ void finalize_kernel(const int* __restrict__ counts, const float* __restrict__ sums,
                                float* __restrict__ out4) {
  __shared__ float sU[1024], sE[1024];
  int t = threadIdx.x;
  int c = counts[t];
  float p = (float)c / (float)NTOK;
  sU[t] = (c >= 1) ? 1.f : 0.f;
  sE[t] = p * logf(p + 1e-10f);
  __syncthreads();
  for (int o = 512; o > 0; o >>= 1) {
    if (t < o) { sU[t] += sU[t + o]; sE[t] += sE[t + o]; }
    __syncthreads();
  }
  if (t == 0) {
    out4[0] = sU[0];                       // used_curr
    out4[1] = expf(-sE[0]);                // perplexity
    float inv = 1.0f / (float)NELEM;
    out4[2] = sums[0] * inv;               // commitment_loss
    out4[3] = sums[1] * inv;               // predictive_loss
  }
}

// ---------------- launch ----------------

extern "C" void kernel_launch(void* const* d_in, const int* in_sizes, int n_in,
                              void* d_out, int out_size, void* d_ws, size_t ws_size,
                              hipStream_t stream) {
  const float* x     = (const float*)d_in[0];
  const float* emb   = (const float*)d_in[1];
  const float* wctx  = (const float*)d_in[2];
  const float* w1    = (const float*)d_in[3];
  const float* a1    = (const float*)d_in[4];
  const float* w2    = (const float*)d_in[5];
  const float* a2    = (const float*)d_in[6];
  const float* noise = (const float*)d_in[7];
  const int*   epo   = (const int*)d_in[8];
  float* out = (float*)d_out;
  char*  ws  = (char*)d_ws;

  // workspace layout (bytes), total ~102.9 MB
  float*          scale   = (float*)(ws + 0);
  float*          sums    = (float*)(ws + 256);
  int*            counts  = (int*)(ws + 512);
  float*          embsq   = (float*)(ws + 4608);
  int*            indices = (int*)(ws + 8704);
  unsigned short* ctxnT   = (unsigned short*)(ws + 139776);     // [B][TCP][D] bf16 (transposed)
  float*          ctxout  = (float*)(ws + 17048064);            // [B][D][T] f32
  unsigned short* curx    = (unsigned short*)(ws + 50602496);   // [B*T][D] bf16
  unsigned short* wctxr   = (unsigned short*)(ws + 67379712);   // [7][D][D] bf16
  unsigned short* ctxT    = (unsigned short*)(ws + 68297216);   // [B*T][D] bf16
  unsigned short* xbf     = (unsigned short*)(ws + 85074432);   // [B*T][D] bf16
  unsigned short* embbf   = (unsigned short*)(ws + 101851648);  // [M][D] bf16
  unsigned short* w1bf    = (unsigned short*)(ws + 102375936);  // [D][2D] bf16
  unsigned short* w2bf    = (unsigned short*)(ws + 102638080);  // [D][2D] bf16

  init_kernel<<<(M_ + 255) / 256, 256, 0, stream>>>(counts, sums);
  scale_kernel<<<B_, 256, 0, stream>>>(x, scale);
  embsq_kernel<<<(M_ + 255) / 256, 256, 0, stream>>>(emb, embsq);
  pack_wctx_kernel<<<(K_ * D_ * D_ + 255) / 256, 256, 0, stream>>>(wctx, wctxr);
  noisy_kernel<<<(B_ * TCP_ * D_ + 255) / 256, 256, 0, stream>>>(x, noise, scale, epo, ctxnT);
  tobf_kernel<<<(NELEM + 255) / 256, 256, 0, stream>>>(x, xbf, NELEM);
  tobf_kernel<<<(M_ * D_ + 255) / 256, 256, 0, stream>>>(emb, embbf, M_ * D_);
  tobf_kernel<<<(D_ * 2 * D_ + 255) / 256, 256, 0, stream>>>(w1, w1bf, D_ * 2 * D_);
  tobf_kernel<<<(D_ * 2 * D_ + 255) / 256, 256, 0, stream>>>(w2, w2bf, D_ * 2 * D_);

  ctx_conv_kernel<<<dim3(B_, D_ / 128, T_ / 64), 128, 0, stream>>>(ctxnT, wctxr, ctxout, ctxT);
  fuse1_kernel<<<dim3(B_, D_ / 128, T_ / 64), 128, 0, stream>>>(xbf, w1bf, a1, ctxT, curx);
  vq_kernel<<<NTOK / 128, 128, 0, stream>>>(curx, embbf, embsq, indices, counts);
  fuse2_kernel<<<dim3(B_, D_ / 128, T_ / 64), 128, 0, stream>>>(embbf, w2bf, a2, ctxT, indices, out);

  loss_kernel<<<2048, 256, 0, stream>>>(x, ctxout, emb, indices, sums);
  idx2f_kernel<<<(NTOK + 255) / 256, 256, 0, stream>>>(indices, out + NELEM);
  finalize_kernel<<<1, 1024, 0, stream>>>(counts, sums, out + NELEM + NTOK);
}
